// MultiHeadAttention_11330123727165
// MI455X (gfx1250) — matrix operand order repaired and checked by reading, not compiled, not run on hardware
//
#include <hip/hip_runtime.h>
#include <hip/hip_bf16.h>
#include <stdint.h>

// ---------------------------------------------------------------------------
// Problem constants
// ---------------------------------------------------------------------------
constexpr int BS   = 2;
constexpr int SEQ  = 2048;
constexpr int EMB  = 1024;
constexpr int NH   = 16;
constexpr int DKH  = 64;          // EMB / NH
constexpr int MTOT = BS * SEQ;    // 4096 rows in every GEMM

// GEMM tiling
constexpr int BM = 128, BN = 128, BK = 32;
constexpr int NKT = EMB / BK;     // 32 K-tiles
constexpr int LDSS = BK + 8;      // LDS row stride (bf16) = 40 -> 80B, 16B aligned

typedef __attribute__((ext_vector_type(16))) __bf16 v16bf;
typedef __attribute__((ext_vector_type(4)))  __bf16 v4bf;
typedef __attribute__((ext_vector_type(8)))  float  v8f;
typedef __attribute__((ext_vector_type(4)))  float  v4f;
typedef __attribute__((ext_vector_type(4)))  unsigned int v4u;

// ---------------------------------------------------------------------------
// Helpers
// ---------------------------------------------------------------------------
__device__ inline __bf16 f2bf(float x) {
    union { float f; unsigned u; } v; v.f = x;
    unsigned r = v.u + 0x7FFFu + ((v.u >> 16) & 1u);   // RNE
    unsigned short h = (unsigned short)(r >> 16);
    __bf16 o;
    __builtin_memcpy(&o, &h, 2);
    return o;
}

// A-matrix fragment (16x32 bf16): two 16B runs per lane
__device__ inline v16bf load_frag_a(const __bf16* p_lo, const __bf16* p_hi) {
    union { v16bf v; v4u u[2]; } f;
    f.u[0] = *reinterpret_cast<const v4u*>(p_lo);
    f.u[1] = *reinterpret_cast<const v4u*>(p_hi);
    return f.v;
}

// B-matrix fragment (32x16 bf16): one contiguous 32B run per lane
__device__ inline v16bf load_frag_b(const __bf16* p) {
    union { v16bf v; v4u u[2]; } f;
    f.u[0] = *reinterpret_cast<const v4u*>(p);
    f.u[1] = *reinterpret_cast<const v4u*>(p + 8);
    return f.v;
}

__device__ inline v8f wmma_bf16(v16bf a, v16bf b, v8f c) {
    return __builtin_amdgcn_wmma_f32_16x16x32_bf16(
        /*neg_a=*/false, a, /*neg_b=*/false, b,
        /*c_mod=*/(short)0, c, /*reuse_a=*/false, /*reuse_b=*/false);
}

// CDNA5 async copy: 16 bytes global -> LDS per lane, tracked by ASYNCcnt.
__device__ inline void async_b128(unsigned lds_off, const __bf16* g) {
    unsigned long long ga = (unsigned long long)(uintptr_t)g;
    asm volatile("global_load_async_to_lds_b128 %0, %1, off"
                 :: "v"(lds_off), "v"(ga) : "memory");
}
__device__ inline void wait_async0() {
    asm volatile("s_wait_asynccnt 0x0" ::: "memory");
}

// ---------------------------------------------------------------------------
// fp32 -> bf16 conversion pass (one-time; keeps GEMM hot loop convert-free)
// ---------------------------------------------------------------------------
__global__ __launch_bounds__(256)
void cvt_bf16_k(const float* __restrict__ in, __bf16* __restrict__ out) {
    const size_t i = ((size_t)blockIdx.x * 256 + threadIdx.x) * 4;
    v4f x = *reinterpret_cast<const v4f*>(in + i);
    union { v4bf v; unsigned long long u; } o;
    for (int j = 0; j < 4; ++j) o.v[j] = f2bf(x[j]);
    *reinterpret_cast<unsigned long long*>(out + i) = o.u;
}

// ---------------------------------------------------------------------------
// GEMM:  C[M,N] = A[M,K] @ W[N,K]^T    (M=4096, N=K=1024, all bf16 inputs)
// Double-buffered LDS filled with global_load_async_to_lds_b128.
// MODE 0: out bf16, head-split [B,H,S,DK]
// MODE 1: out bf16, transposed [B,H,DK,S]   (for V)
// MODE 2: out fp32 to d_out, + bias[n]
// ---------------------------------------------------------------------------
template<int MODE>
__global__ __launch_bounds__(256)
void gemm_wmma_k(const __bf16* __restrict__ A,
                 const __bf16* __restrict__ W,
                 const float*  __restrict__ bias,
                 void* __restrict__ Out) {
    __shared__ __align__(16) __bf16 Asm[2][BM * LDSS];
    __shared__ __align__(16) __bf16 Bsm[2][BN * LDSS];

    const int tid  = threadIdx.x;
    const int wid  = tid >> 5;
    const int lane = tid & 31;
    const int half = lane >> 4;
    const int l16  = lane & 15;
    const int wm   = wid & 3;        // 4 wave-rows (32 rows each)
    const int wn   = wid >> 2;       // 2 wave-cols (64 cols each)
    const int bm0  = blockIdx.y * BM;
    const int bn0  = blockIdx.x * BN;

    // async staging pattern: each thread copies one 16B chunk per 64-row slab
    const int chunk = tid & 3;       // 4 chunks of 8 bf16 per 32-elem row
    const int rowI  = tid >> 2;      // 0..63
    const unsigned aBase[2] = { (unsigned)(uintptr_t)&Asm[0][0],
                                (unsigned)(uintptr_t)&Asm[1][0] };
    const unsigned bBase[2] = { (unsigned)(uintptr_t)&Bsm[0][0],
                                (unsigned)(uintptr_t)&Bsm[1][0] };

    auto issue_tile = [&](int kt, int buf) {
        const int k0 = kt * BK;
        for (int p = 0; p < 2; ++p) {
            const int r = p * 64 + rowI;
            const unsigned loff = (unsigned)((r * LDSS + chunk * 8) * 2);
            async_b128(aBase[buf] + loff, A + (size_t)(bm0 + r) * EMB + k0 + chunk * 8);
            async_b128(bBase[buf] + loff, W + (size_t)(bn0 + r) * EMB + k0 + chunk * 8);
        }
    };

    const v8f vzero = {0,0,0,0,0,0,0,0};
    v8f acc[2][4];
    for (int i = 0; i < 2; ++i)
        for (int j = 0; j < 4; ++j) acc[i][j] = vzero;

    issue_tile(0, 0);

    for (int kt = 0; kt < NKT; ++kt) {
        const int cur = kt & 1;
        wait_async0();        // own copies for buf[cur] done
        __syncthreads();      // everyone's copies done
        if (kt + 1 < NKT) issue_tile(kt + 1, cur ^ 1);   // overlap with WMMAs

        v16bf af[2], bfr[4];
        for (int mt = 0; mt < 2; ++mt) {
            const __bf16* row = &Asm[cur][(wm * 32 + mt * 16 + l16) * LDSS];
            af[mt] = load_frag_a(row + half * 16, row + half * 16 + 8);
        }
        for (int nt = 0; nt < 4; ++nt) {
            const __bf16* row = &Bsm[cur][(wn * 64 + nt * 16 + l16) * LDSS];
            bfr[nt] = load_frag_b(row + half * 16);
        }
        for (int mt = 0; mt < 2; ++mt)
            for (int nt = 0; nt < 4; ++nt)
                acc[mt][nt] = wmma_bf16(af[mt], bfr[nt], acc[mt][nt]);
        __syncthreads();      // readers done before buf[cur] is refilled
    }

    // ---- store ----
    for (int mt = 0; mt < 2; ++mt) {
        for (int nt = 0; nt < 4; ++nt) {
            for (int r = 0; r < 8; ++r) {
                const int m = bm0 + wm * 32 + mt * 16 + r + 8 * half;
                const int n = bn0 + wn * 64 + nt * 16 + l16;
                const float v = acc[mt][nt][r];
                if (MODE == 2) {
                    ((float*)Out)[(size_t)m * EMB + n] = v + bias[n];
                } else {
                    const int b = m / SEQ, s = m % SEQ;
                    const int h = n / DKH, dk = n % DKH;
                    __bf16* O = (__bf16*)Out;
                    if (MODE == 0)
                        O[(((size_t)(b * NH + h) * SEQ + s) * DKH + dk)] = f2bf(v);
                    else
                        O[(((size_t)(b * NH + h) * DKH + dk) * SEQ + s)] = f2bf(v);
                }
            }
        }
    }
}

// ---------------------------------------------------------------------------
// Flash attention. Block = 128 threads = 4 waves; each wave owns a 16-row
// query tile of one (b,h). 32 keys per iteration -> 8 WMMAs / iter.
// Q,K: [B,H,S,DK] bf16;  Vt: [B,H,DK,S] bf16;  Oattn: [B,S,E] bf16
// ---------------------------------------------------------------------------
__global__ __launch_bounds__(128)
void attn_wmma_k(const __bf16* __restrict__ Q,
                 const __bf16* __restrict__ K,
                 const __bf16* __restrict__ Vt,
                 const float*  __restrict__ rel_table,
                 __bf16* __restrict__ Oattn) {
    __shared__ __align__(16) float  bias_s[2 * SEQ - 1];   // 4095 floats for head h
    __shared__ __align__(16) __bf16 Psm[4][16 * 40];       // per-wave P staging

    const int qblocks = SEQ / 64;                 // 32
    const int bh = blockIdx.x / qblocks;          // b*NH + h
    const int qb = blockIdx.x % qblocks;
    const int b  = bh / NH, h = bh % NH;

    const int tid  = threadIdx.x;
    const int wid  = tid >> 5;
    const int lane = tid & 31;
    const int half = lane >> 4;
    const int l16  = lane & 15;

    for (int i = tid; i < 2 * SEQ - 1; i += 128)
        bias_s[i] = rel_table[(size_t)i * NH + h];
    __syncthreads();

    const int q0 = qb * 64 + wid * 16;
    const __bf16* Qb = Q  + (size_t)bh * SEQ * DKH;
    const __bf16* Kb = K  + (size_t)bh * SEQ * DKH;
    const __bf16* Vb = Vt + (size_t)bh * DKH * SEQ;

    // Q A-fragments (dk 0..31 and 32..63)
    v16bf aq[2];
    {
        const __bf16* qrow = Qb + (size_t)(q0 + l16) * DKH;
        aq[0] = load_frag_a(qrow + half * 16,      qrow + half * 16 + 8);
        aq[1] = load_frag_a(qrow + 32 + half * 16, qrow + 40 + half * 16);
    }

    const v8f vzero = {0,0,0,0,0,0,0,0};
    v8f acc[4]; for (int nt = 0; nt < 4; ++nt) acc[nt] = vzero;
    float mi[8], li[8];
    for (int r = 0; r < 8; ++r) { mi[r] = -__builtin_inff(); li[r] = 0.f; }

    __bf16* Pw = &Psm[wid][0];

    for (int kb = 0; kb < SEQ; kb += 32) {
        // ---- scores: two 16x16 tiles via 4 WMMAs ----
        v8f s[2];
        for (int t = 0; t < 2; ++t) {
            const __bf16* krow = Kb + (size_t)(kb + t * 16 + l16) * DKH;
            v16bf kf0 = load_frag_b(krow + half * 16);
            v16bf kf1 = load_frag_b(krow + 32 + half * 16);
            v8f c = vzero;
            c = wmma_bf16(aq[0], kf0, c);
            c = wmma_bf16(aq[1], kf1, c);
            s[t] = c;
        }

        // ---- bias + online softmax (per row r, 16-lane butterflies) ----
        const float sc = 0.125f;  // 1/sqrt(64)
        float p0[8], p1[8];
        for (int r = 0; r < 8; ++r) {
            const int q = q0 + r + 8 * half;
            const int key0 = kb + l16;
            float s0 = s[0][r] * sc + bias_s[key0 - q + (SEQ - 1)];
            float s1 = s[1][r] * sc + bias_s[key0 + 16 - q + (SEQ - 1)];
            float rm = fmaxf(s0, s1);
            for (int msk = 1; msk < 16; msk <<= 1)
                rm = fmaxf(rm, __shfl_xor(rm, msk, 32));
            const float mnew = fmaxf(mi[r], rm);
            const float e0 = __expf(s0 - mnew);
            const float e1 = __expf(s1 - mnew);
            float rs = e0 + e1;
            for (int msk = 1; msk < 16; msk <<= 1)
                rs += __shfl_xor(rs, msk, 32);
            const float sold = __expf(mi[r] - mnew);
            li[r] = li[r] * sold + rs;
            mi[r] = mnew;
            for (int nt = 0; nt < 4; ++nt) acc[nt][r] *= sold;
            p0[r] = e0; p1[r] = e1;
        }

        // ---- C-layout -> A-layout via wave-private LDS ----
        for (int r = 0; r < 8; ++r) {
            const int row = r + 8 * half;
            Pw[row * 40 + l16]      = f2bf(p0[r]);
            Pw[row * 40 + 16 + l16] = f2bf(p1[r]);
        }
        const __bf16* prow = &Pw[l16 * 40];
        v16bf pa = load_frag_a(prow + half * 16, prow + half * 16 + 8);

        // ---- O += P @ V  (V transposed: contiguous B-frags) ----
        for (int nt = 0; nt < 4; ++nt) {
            const __bf16* vrow = Vb + (size_t)(nt * 16 + l16) * SEQ + kb;
            v16bf vf = load_frag_b(vrow + half * 16);
            acc[nt] = wmma_bf16(pa, vf, acc[nt]);
        }
    }

    // ---- normalize + store merged-heads bf16 [B,S,E] ----
    for (int r = 0; r < 8; ++r) {
        const float inv = 1.0f / li[r];
        const int q = q0 + r + 8 * half;
        for (int nt = 0; nt < 4; ++nt) {
            Oattn[((size_t)b * SEQ + q) * EMB + h * DKH + nt * 16 + l16] =
                f2bf(acc[nt][r] * inv);
        }
    }
}

// ---------------------------------------------------------------------------
// Launcher
// ---------------------------------------------------------------------------
extern "C" void kernel_launch(void* const* d_in, const int* in_sizes, int n_in,
                              void* d_out, int out_size, void* d_ws, size_t ws_size,
                              hipStream_t stream) {
    (void)in_sizes; (void)n_in; (void)out_size; (void)ws_size;

    const float* key_i  = (const float*)d_in[0];
    const float* query  = (const float*)d_in[1];
    const float* value  = (const float*)d_in[2];
    const float* wq     = (const float*)d_in[3];
    const float* wk     = (const float*)d_in[4];
    const float* wv     = (const float*)d_in[5];
    const float* fc_w   = (const float*)d_in[6];
    const float* fc_b   = (const float*)d_in[7];
    const float* relt   = (const float*)d_in[8];
    float* out = (float*)d_out;

    const size_t nAct = (size_t)MTOT * EMB;   // 4M elems
    const size_t nWgt = (size_t)EMB * EMB;    // 1M elems
    __bf16* qa   = (__bf16*)d_ws;             // bf16 activations
    __bf16* ka   = qa  + nAct;
    __bf16* va   = ka  + nAct;
    __bf16* wqb  = va  + nAct;                // bf16 weights
    __bf16* wkb  = wqb + nWgt;
    __bf16* wvb  = wkb + nWgt;
    __bf16* wfb  = wvb + nWgt;
    __bf16* qh   = wfb + nWgt;                // projected, head-split
    __bf16* kh   = qh  + nAct;
    __bf16* vth  = kh  + nAct;
    __bf16* o_ws = qa;                        // alias: qa is dead after Q-proj

    // 1) fp32 -> bf16 conversions
    const int actBlocks = (int)(nAct / (256 * 4));   // 4096
    const int wgtBlocks = (int)(nWgt / (256 * 4));   // 1024
    cvt_bf16_k<<<actBlocks, 256, 0, stream>>>(query, qa);
    cvt_bf16_k<<<actBlocks, 256, 0, stream>>>(key_i, ka);
    cvt_bf16_k<<<actBlocks, 256, 0, stream>>>(value, va);
    cvt_bf16_k<<<wgtBlocks, 256, 0, stream>>>(wq,   wqb);
    cvt_bf16_k<<<wgtBlocks, 256, 0, stream>>>(wk,   wkb);
    cvt_bf16_k<<<wgtBlocks, 256, 0, stream>>>(wv,   wvb);
    cvt_bf16_k<<<wgtBlocks, 256, 0, stream>>>(fc_w, wfb);

    // 2) projections (WMMA, async-LDS staged)
    dim3 gg(EMB / BN, MTOT / BM);             // (8, 32)
    gemm_wmma_k<0><<<gg, 256, 0, stream>>>(qa, wqb, nullptr, qh);
    gemm_wmma_k<0><<<gg, 256, 0, stream>>>(ka, wkb, nullptr, kh);
    gemm_wmma_k<1><<<gg, 256, 0, stream>>>(va, wvb, nullptr, vth);

    // 3) flash attention with relative-position bias
    attn_wmma_k<<<BS * NH * (SEQ / 64), 128, 0, stream>>>(qh, kh, vth, relt, o_ws);

    // 4) output projection + bias (fp32 out)
    gemm_wmma_k<2><<<gg, 256, 0, stream>>>(o_ws, wfb, fc_b, out);
}
